// RowSharNotReducRef_32933809226154
// MI455X (gfx1250) — matrix operand order repaired and checked by reading, not compiled, not run on hardware
//
#include <hip/hip_runtime.h>
#include <hip/hip_bf16.h>
#include <math.h>

typedef __bf16 bf16;
typedef __attribute__((ext_vector_type(16))) __bf16 v16bf;
typedef __attribute__((ext_vector_type(8)))  __bf16 v8bf;
typedef __attribute__((ext_vector_type(8)))  float  v8f;

// ---- problem constants (from reference) ----
constexpr int B_   = 32;
constexpr int DF   = 8;
constexpr int H_   = 144;
constexpr int W_   = 144;
constexpr int OFF_ = 2;
constexpr int Kw   = 5;           // window size 2*OFF+1
constexpr int DS   = 512;
constexpr int DT   = 1024;
constexpr int NC   = 6;
constexpr int HEADS= 16;
constexpr int DH   = 64;
constexpr int MLP  = 2048;
constexpr int CIN  = DF * W_;     // 1152
constexpr int ITC  = DF * H_ * Kw;// 5760
constexpr int WP   = W_ + 2*OFF_ + Kw; // 153 (padded width)
constexpr int INNER= HEADS * DH;  // 1024
constexpr float EPS = 1e-5f;
constexpr float THR = 0.3f;

// =====================================================================
// Generic batched bf16 WMMA GEMM:  C[z] = A[z] (MxK, rm) * Bw[z]^T (NxK, rm)
// One wave computes a 32x32 macro-tile (2x2 fragments of 16x16) so each
// fragment load feeds two v_wmma_f32_16x16x32_bf16 ops (2 b128 loads per
// WMMA instead of 4).  Edge macro-tiles clamp the second fragment to the
// last valid 16-tile (wave-uniform redundant compute) and guard stores.
// Fused epilogue:
//   v = acc * scaleM[m] + biasM[m] + biasN[n]; gelu?; += resid; f32/bf16
// Batch index z = zo*innerCnt + zi with independent strides per operand.
// =====================================================================
__device__ __forceinline__ void frag_epilogue(
    v8f acc, int tm, int tn, int l, int hf,
    float* Cf, bf16* Cbf,
    const float* scaleM, const float* biasM, const float* biasN,
    const float* resid, long cOff, long pOff,
    int ldc, int transposeStore, int gelu)
{
    int n_g = tn * 16 + l;
    float bN = biasN ? biasN[n_g] : 0.f;
#pragma unroll
    for (int j = 0; j < 8; ++j) {
        int m_g = tm * 16 + hf * 8 + j;   // C layout: VGPR j -> M = j + 8*half
        float v = acc[j];
        if (scaleM) v *= scaleM[pOff + m_g];
        if (biasM)  v += biasM[pOff + m_g];
        v += bN;
        if (gelu)   v = 0.5f * v * (1.f + erff(v * 0.70710678118f));
        long idx = cOff + (transposeStore ? ((long)n_g * ldc + m_g)
                                          : ((long)m_g * ldc + n_g));
        if (resid) v += resid[idx];
        if (Cf)  Cf[idx]  = v;
        if (Cbf) Cbf[idx] = (bf16)v;
    }
}

__global__ void __launch_bounds__(256) k_wmma_gemm(
    const bf16* __restrict__ A, const bf16* __restrict__ Bw,
    float* Cf, bf16* Cbf,
    const float* scaleM, const float* biasM, const float* biasN,
    const float* resid,
    int M, int N, int Kd, int ldc,
    int transposeStore, int gelu, int innerCnt,
    long sAo, long sAi, long sBo, long sBi,
    long sCo, long sCi, long sPi)
{
    int z  = blockIdx.z;
    int zo = z / innerCnt, zi = z % innerCnt;
    A  += zo * sAo + zi * sAi;
    Bw += zo * sBo + zi * sBi;
    long cOff = zo * sCo + zi * sCi;
    long pOff = (long)zi * sPi;

    int tilesN  = N >> 4;
    int tilesM  = M >> 4;
    int mtilesN = (tilesN + 1) >> 1;       // macro tiles (32-wide)
    int mtilesM = (tilesM + 1) >> 1;
    int wid = blockIdx.x * (blockDim.x >> 5) + (threadIdx.x >> 5);
    if (wid >= mtilesM * mtilesN) return;  // wave-uniform: EXEC stays all-1s
    int tm0 = (wid / mtilesN) * 2, tn0 = (wid % mtilesN) * 2;
    int tm1 = tm0 + 1; if (tm1 >= tilesM) tm1 = tilesM - 1;  // clamp edges
    int tn1 = tn0 + 1; if (tn1 >= tilesN) tn1 = tilesN - 1;
    int lane = threadIdx.x & 31;
    int hf = lane >> 4;        // lane half selector
    int l  = lane & 15;

    const bf16* ar0 = A  + (size_t)(tm0 * 16 + l) * Kd;
    const bf16* ar1 = A  + (size_t)(tm1 * 16 + l) * Kd;
    const bf16* br0 = Bw + (size_t)(tn0 * 16 + l) * Kd;
    const bf16* br1 = Bw + (size_t)(tn1 * 16 + l) * Kd;

    v8f acc00 = {0.f,0.f,0.f,0.f,0.f,0.f,0.f,0.f};
    v8f acc01 = acc00, acc10 = acc00, acc11 = acc00;

    for (int kb = 0; kb < Kd; kb += 32) {
        // A 16x32 bf16 fragment: lanes 0-15 rows, halves select K sub-blocks
        v8bf a0lo = *(const v8bf*)(ar0 + kb + hf * 8);
        v8bf a0hi = *(const v8bf*)(ar0 + kb + 16 + hf * 8);
        v8bf a1lo = *(const v8bf*)(ar1 + kb + hf * 8);
        v8bf a1hi = *(const v8bf*)(ar1 + kb + 16 + hf * 8);
        // B 32x16 bf16 fragment: lane = column n, halves select K sub-blocks
        v8bf b0lo = *(const v8bf*)(br0 + kb + hf * 8);
        v8bf b0hi = *(const v8bf*)(br0 + kb + 16 + hf * 8);
        v8bf b1lo = *(const v8bf*)(br1 + kb + hf * 8);
        v8bf b1hi = *(const v8bf*)(br1 + kb + 16 + hf * 8);
        v16bf av0 = __builtin_shufflevector(a0lo, a0hi, 0,1,2,3,4,5,6,7,8,9,10,11,12,13,14,15);
        v16bf av1 = __builtin_shufflevector(a1lo, a1hi, 0,1,2,3,4,5,6,7,8,9,10,11,12,13,14,15);
        v16bf bv0 = __builtin_shufflevector(b0lo, b0hi, 0,1,2,3,4,5,6,7,8,9,10,11,12,13,14,15);
        v16bf bv1 = __builtin_shufflevector(b1lo, b1hi, 0,1,2,3,4,5,6,7,8,9,10,11,12,13,14,15);
        acc00 = __builtin_amdgcn_wmma_f32_16x16x32_bf16(false, av0, false, bv0, (short)0, acc00, false, false);
        acc01 = __builtin_amdgcn_wmma_f32_16x16x32_bf16(false, av0, false, bv1, (short)0, acc01, false, false);
        acc10 = __builtin_amdgcn_wmma_f32_16x16x32_bf16(false, av1, false, bv0, (short)0, acc10, false, false);
        acc11 = __builtin_amdgcn_wmma_f32_16x16x32_bf16(false, av1, false, bv1, (short)0, acc11, false, false);
    }

    bool mOk = (tm0 + 1 < tilesM);
    bool nOk = (tn0 + 1 < tilesN);
    frag_epilogue(acc00, tm0, tn0, l, hf, Cf, Cbf, scaleM, biasM, biasN,
                  resid, cOff, pOff, ldc, transposeStore, gelu);
    if (nOk)
        frag_epilogue(acc01, tm0, tn1, l, hf, Cf, Cbf, scaleM, biasM, biasN,
                      resid, cOff, pOff, ldc, transposeStore, gelu);
    if (mOk)
        frag_epilogue(acc10, tm1, tn0, l, hf, Cf, Cbf, scaleM, biasM, biasN,
                      resid, cOff, pOff, ldc, transposeStore, gelu);
    if (mOk && nOk)
        frag_epilogue(acc11, tm1, tn1, l, hf, Cf, Cbf, scaleM, biasM, biasN,
                      resid, cOff, pOff, ldc, transposeStore, gelu);
}

// ---- small glue kernels ----

__global__ void k_cvt_bf16(bf16* dst, const float* src, int n) {
    int i = blockIdx.x * blockDim.x + threadIdx.x;
    if (i < n) dst[i] = (bf16)src[i];
}

// folded affine for heads "batch norm": alpha = g/sqrt(vr+eps), beta = (b1-mu)*alpha + bt
__global__ void k_affine(float* alpha, float* beta, const float* b1, const float* g,
                         const float* bt, const float* mu, const float* vr, int n) {
    int i = blockIdx.x * blockDim.x + threadIdx.x;
    if (i < n) {
        float a = g[i] * rsqrtf(vr[i] + EPS);
        alpha[i] = a;
        beta[i]  = (b1[i] - mu[i]) * a + bt[i];
    }
}

// Xt[b][h][f*W+w] = x[b][f][h][w]   (K-contiguous B operand for heads GEMM1)
__global__ void k_build_xt(bf16* xt, const float* x) {
    int i = blockIdx.x * blockDim.x + threadIdx.x;
    if (i >= B_ * H_ * CIN) return;
    int c = i % CIN; int t = i / CIN;
    int h = t % H_;  int b = t / H_;
    int f = c / W_;  int w = c % W_;
    xt[i] = (bf16)x[((size_t)(b * DF + f) * H_ + h) * W_ + w];
}

// Round-2 version: x2 = pad[:, :, :, OFF:OFF+W]
__global__ void k_build_xt2(bf16* xt, const float* pad) {
    int i = blockIdx.x * blockDim.x + threadIdx.x;
    if (i >= B_ * H_ * CIN) return;
    int c = i % CIN; int t = i / CIN;
    int h = t % H_;  int b = t / H_;
    int f = c / W_;  int w = c % W_;
    xt[i] = (bf16)pad[((size_t)(b * DF + f) * H_ + h) * WP + OFF_ + w];
}

// ext head: O=2 logits + softmax, straight to d_out region
__global__ void k_ext_head(float* out, const bf16* hn, const float* w2, const float* b2) {
    int i = blockIdx.x * blockDim.x + threadIdx.x; // over B*NC*H
    if (i >= B_ * NC * H_) return;
    int h = i % H_; int bk = i / H_; int k = bk % NC;
    const bf16* hr = hn + (size_t)bk * H_ * DS + (size_t)h * DS;
    const float* w0 = w2 + ((size_t)k * 2 + 0) * DS;
    const float* w1 = w2 + ((size_t)k * 2 + 1) * DS;
    float a0 = b2[k * 2 + 0], a1 = b2[k * 2 + 1];
    for (int s = 0; s < DS; ++s) {
        float v = (float)hr[s];
        a0 += w0[s] * v;
        a1 += w1[s] * v;
    }
    float mx = fmaxf(a0, a1);
    float e0 = __expf(a0 - mx), e1 = __expf(a1 - mx);
    float inv = 1.f / (e0 + e1);
    out[(size_t)i * 2 + 0] = e0 * inv;
    out[(size_t)i * 2 + 1] = e1 * inv;
}

__global__ void k_exist(int* exist, const float* extOut) {
    int i = blockIdx.x * blockDim.x + threadIdx.x;
    if (i >= B_ * NC) return;
    float s = 0.f;
    for (int h = 0; h < H_; ++h) s += extOut[((size_t)i * H_ + h) * 2];
    exist[i] = (s * (1.f / H_) > THR) ? 1 : 0;
}

// in-place softmax over O per row + first-max argmax -> corr (one wave per row)
__global__ void k_cls_softmax(float* logits, int* corr, int O) {
    int wid  = (blockIdx.x * blockDim.x + threadIdx.x) >> 5;
    int lane = threadIdx.x & 31;
    if (wid >= B_ * NC * H_) return;
    float* row = logits + (size_t)wid * O;
    float mx = -1e30f; int mi = 0x7fffffff;
    for (int o = lane; o < O; o += 32) {
        float v = row[o];
        if (v > mx) { mx = v; mi = o; }
    }
    for (int off = 16; off; off >>= 1) {
        float om = __shfl_xor(mx, off, 32);
        int   oi = __shfl_xor(mi, off, 32);
        if (om > mx || (om == mx && oi < mi)) { mx = om; mi = oi; }
    }
    float sum = 0.f;
    for (int o = lane; o < O; o += 32) sum += __expf(row[o] - mx);
    for (int off = 16; off; off >>= 1) sum += __shfl_xor(sum, off, 32);
    float inv = 1.f / sum;
    for (int o = lane; o < O; o += 32) row[o] = __expf(row[o] - mx) * inv;
    if (lane == 0) corr[wid] = mi;
}

__global__ void k_build_pad(float* pad, const float* x) {
    int i = blockIdx.x * blockDim.x + threadIdx.x;
    if (i >= B_ * DF * H_ * WP) return;
    int wp = i % WP; int t = i / WP;       // t = (b*DF+f)*H + h
    int w = wp - OFF_;
    pad[i] = (w >= 0 && w < W_) ? x[(size_t)t * W_ + w] : 0.f;
}

// win[b*NC+c][f*H*K + h*K + kk] = pad[b][f][h][corr[b,c,h]+kk]  (bf16 for tok GEMM)
__global__ void k_gather_win(bf16* win, const float* pad, const int* corr) {
    int i = blockIdx.x * blockDim.x + threadIdx.x;
    if (i >= B_ * NC * ITC) return;
    int itc = i % ITC; int bc = i / ITC;
    int b = bc / NC;
    int kk = itc % Kw; int t = itc / Kw;
    int h = t % H_;    int f = t / H_;
    int col = corr[bc * H_ + h] + kk;
    win[i] = (bf16)pad[((size_t)(b * DF + f) * H_ + h) * WP + col];
}

__global__ void k_add_emb(float* tok, const float* emb) {
    int i = blockIdx.x * blockDim.x + threadIdx.x;
    if (i >= B_ * NC * DT) return;
    int d = i % DT; int c = (i / DT) % NC;
    tok[i] += emb[c * DT + d];
}

// LayerNorm over last dim -> bf16 (block per row, LDS reductions)
__global__ void k_layernorm(bf16* out, const float* in, const float* g, const float* bt, int D) {
    int row = blockIdx.x;
    const float* xr = in + (size_t)row * D;
    __shared__ float red[256];
    float s = 0.f;
    for (int d = threadIdx.x; d < D; d += 256) s += xr[d];
    red[threadIdx.x] = s; __syncthreads();
    for (int off = 128; off; off >>= 1) {
        if (threadIdx.x < off) red[threadIdx.x] += red[threadIdx.x + off];
        __syncthreads();
    }
    float mean = red[0] / D; __syncthreads();
    float q = 0.f;
    for (int d = threadIdx.x; d < D; d += 256) { float dv = xr[d] - mean; q += dv * dv; }
    red[threadIdx.x] = q; __syncthreads();
    for (int off = 128; off; off >>= 1) {
        if (threadIdx.x < off) red[threadIdx.x] += red[threadIdx.x + off];
        __syncthreads();
    }
    float inv = rsqrtf(red[0] / D + EPS);
    for (int d = threadIdx.x; d < D; d += 256)
        out[(size_t)row * D + d] = (bf16)((xr[d] - mean) * inv * g[d] + bt[d]);
}

// tiny attention over NC=6 tokens: thread per (b, head, n)
__global__ void k_attn(bf16* o, const float* qkv, const int* exist) {
    int t = blockIdx.x * blockDim.x + threadIdx.x;
    if (t >= B_ * HEADS * NC) return;
    int n = t % NC; int r = t / NC;
    int hh = r % HEADS; int b = r / HEADS;
    const float* q = qkv + (size_t)(b * NC + n) * 3 * INNER + hh * DH;
    float s[NC]; float mx = -1e30f;
    for (int m = 0; m < NC; ++m) {
        const float* kp = qkv + (size_t)(b * NC + m) * 3 * INNER + INNER + hh * DH;
        float d = 0.f;
        for (int j = 0; j < DH; ++j) d += q[j] * kp[j];
        d *= 0.125f;                       // DH^-0.5
        if (!exist[b * NC + m]) d = -1e9f;
        s[m] = d; mx = fmaxf(mx, d);
    }
    float sum = 0.f;
    for (int m = 0; m < NC; ++m) { s[m] = __expf(s[m] - mx); sum += s[m]; }
    float inv = 1.f / sum;
    for (int m = 0; m < NC; ++m) s[m] *= inv;
    for (int j = 0; j < DH; ++j) {
        float acc = 0.f;
        for (int m = 0; m < NC; ++m)
            acc += s[m] * qkv[(size_t)(b * NC + m) * 3 * INNER + 2 * INNER + hh * DH + j];
        o[(size_t)(b * NC + n) * INNER + hh * DH + j] = (bf16)acc;
    }
}

// sequential-over-c scatter; each thread owns one (b,f,h) row of pad
__global__ void k_scatter(float* pad, const float* dec, const int* corr, const int* exist) {
    int i = blockIdx.x * blockDim.x + threadIdx.x;
    if (i >= B_ * DF * H_) return;
    int h = i % H_; int t = i / H_;
    int f = t % DF; int b = t / DF;
    if (h == H_ - 1) return;               // hmask: writes would go to trash region
    float* prow = pad + (size_t)i * WP;
    for (int c = 0; c < NC; ++c) {         // serial over c preserves overwrite order
        if (!exist[b * NC + c]) continue;  // non-existing -> trash writes (never read)
        int col = corr[(b * NC + c) * H_ + h];
        const float* dr = dec + (size_t)(b * NC + c) * ITC + ((size_t)f * H_ + h) * Kw;
        for (int kk = 0; kk < Kw; ++kk) prow[col + kk] = dr[kk];
    }
}

// =====================================================================
// Host orchestration
// =====================================================================
static inline void gemm(hipStream_t st,
                        const bf16* A, const bf16* Bw, float* Cf, bf16* Cbf,
                        const float* sM, const float* bM, const float* bN, const float* resid,
                        int M, int N, int Kd, int ldc, int tr, int gelu,
                        int batches, int innerCnt,
                        long sAo, long sAi, long sBo, long sBi, long sCo, long sCi, long sPi)
{
    int mtiles = ((M / 16 + 1) / 2) * ((N / 16 + 1) / 2);  // 32x32 macro tiles
    int blocks = (mtiles + 7) / 8;         // 8 waves per 256-thread block
    dim3 grid(blocks, 1, batches);
    k_wmma_gemm<<<grid, 256, 0, st>>>(A, Bw, Cf, Cbf, sM, bM, bN, resid,
                                      M, N, Kd, ldc, tr, gelu, innerCnt,
                                      sAo, sAi, sBo, sBi, sCo, sCi, sPi);
}

extern "C" void kernel_launch(void* const* d_in, const int* in_sizes, int n_in,
                              void* d_out, int out_size, void* d_ws, size_t ws_size,
                              hipStream_t stream) {
    (void)in_sizes; (void)n_in; (void)out_size; (void)ws_size;

    const float* x     = (const float*)d_in[0];
    const float* emb   = (const float*)d_in[33];
    const float* tok_w = (const float*)d_in[34];
    const float* tok_b = (const float*)d_in[35];
    const float* ln1_g = (const float*)d_in[36];
    const float* ln1_b = (const float*)d_in[37];
    const float* qkv_w = (const float*)d_in[38];
    const float* out_w = (const float*)d_in[39];
    const float* out_b = (const float*)d_in[40];
    const float* ln2_g = (const float*)d_in[41];
    const float* ln2_b = (const float*)d_in[42];
    const float* ff_w1 = (const float*)d_in[43];
    const float* ff_b1 = (const float*)d_in[44];
    const float* ff_w2 = (const float*)d_in[45];
    const float* ff_b2 = (const float*)d_in[46];
    const float* lnf_g = (const float*)d_in[47];
    const float* lnf_b = (const float*)d_in[48];
    const float* dec_w = (const float*)d_in[49];
    const float* dec_b = (const float*)d_in[50];

    float* out = (float*)d_out;
    const size_t EXT_SZ = (size_t)B_ * NC * H_ * 2;
    const size_t CLS_SZ = (size_t)B_ * NC * H_ * W_;
    float* outExt  = out;
    float* outCls  = out + EXT_SZ;
    float* outExt2 = out + EXT_SZ + CLS_SZ;
    float* outCls2 = out + 2 * EXT_SZ + CLS_SZ;

    // ---- workspace bump allocator ----
    char* p = (char*)d_ws;
    auto alloc = [&](size_t bytes) -> void* {
        void* r = p; p += (bytes + 255) & ~(size_t)255; return r;
    };
    bf16* w1A    = (bf16*)alloc((size_t)NC * DS * CIN * 2);   // heads w1 (current round, ext)
    bf16* w1B    = (bf16*)alloc((size_t)NC * DS * CIN * 2);   // heads w1 (current round, cls)
    bf16* w2c    = (bf16*)alloc((size_t)NC * W_ * DS * 2);    // heads cls w2 (current round)
    bf16* tokwb  = (bf16*)alloc((size_t)DT * ITC * 2);
    bf16* qkvwb  = (bf16*)alloc((size_t)3 * INNER * DT * 2);
    bf16* outwb  = (bf16*)alloc((size_t)DT * INNER * 2);
    bf16* ffw1b  = (bf16*)alloc((size_t)MLP * DT * 2);
    bf16* ffw2b  = (bf16*)alloc((size_t)DT * MLP * 2);
    bf16* decwb  = (bf16*)alloc((size_t)ITC * DT * 2);
    float* alphaB= (float*)alloc((size_t)4 * NC * DS * 4);
    float* betaB = (float*)alloc((size_t)4 * NC * DS * 4);
    bf16* xtb    = (bf16*)alloc((size_t)B_ * H_ * CIN * 2);
    bf16* hn     = (bf16*)alloc((size_t)B_ * NC * H_ * DS * 2);
    float* pad   = (float*)alloc((size_t)B_ * DF * H_ * WP * 4);
    int*  corr   = (int*)alloc((size_t)B_ * NC * H_ * 4);
    int*  exist  = (int*)alloc((size_t)B_ * NC * 4);
    bf16* winb   = (bf16*)alloc((size_t)B_ * NC * ITC * 2);
    float* tok   = (float*)alloc((size_t)B_ * NC * DT * 4);
    float* tok2  = (float*)alloc((size_t)B_ * NC * DT * 4);
    float* tokf  = (float*)alloc((size_t)B_ * NC * DT * 4);
    bf16* ybf    = (bf16*)alloc((size_t)B_ * NC * DT * 2);
    float* qkvb  = (float*)alloc((size_t)B_ * NC * 3 * INNER * 4);
    bf16* attnb  = (bf16*)alloc((size_t)B_ * NC * INNER * 2);
    bf16* g1b    = (bf16*)alloc((size_t)B_ * NC * MLP * 2);
    float* decb  = (float*)alloc((size_t)B_ * NC * ITC * 4);

    auto cvt = [&](bf16* dst, const float* src, size_t n) {
        k_cvt_bf16<<<(unsigned)((n + 255) / 256), 256, 0, stream>>>(dst, src, (int)n);
    };

    // folded affines for all 4 head sets (ext, cls, ext2, cls2)
    const int bases[4] = {1, 9, 17, 25}; // w1,b1,g,bt,mu,vr,w2,b2 per set
    for (int s = 0; s < 4; ++s) {
        int bi = bases[s];
        k_affine<<<(NC * DS + 255) / 256, 256, 0, stream>>>(
            alphaB + (size_t)s * NC * DS, betaB + (size_t)s * NC * DS,
            (const float*)d_in[bi + 1], (const float*)d_in[bi + 2],
            (const float*)d_in[bi + 3], (const float*)d_in[bi + 4],
            (const float*)d_in[bi + 5], NC * DS);
    }

    // transformer weights -> bf16 (once)
    cvt(tokwb, tok_w, (size_t)DT * ITC);
    cvt(qkvwb, qkv_w, (size_t)3 * INNER * DT);
    cvt(outwb, out_w, (size_t)DT * INNER);
    cvt(ffw1b, ff_w1, (size_t)MLP * DT);
    cvt(ffw2b, ff_w2, (size_t)DT * MLP);
    cvt(decwb, dec_w, (size_t)ITC * DT);

    const long sHN = (long)H_ * DS;        // hn per-(b,k) stride
    const int ROWS = B_ * NC * H_;         // 27648

    // ============== Round 1 heads ==============
    cvt(w1A, (const float*)d_in[1], (size_t)NC * DS * CIN);   // ext_w1
    cvt(w1B, (const float*)d_in[9], (size_t)NC * DS * CIN);   // cls_w1
    cvt(w2c, (const float*)d_in[15], (size_t)NC * W_ * DS);   // cls_w2
    k_build_xt<<<(B_ * H_ * CIN + 255) / 256, 256, 0, stream>>>(xtb, x);

    // ext GEMM1: hn[b,k][h][s] = norm(w1 @ Xt)
    gemm(stream, w1A, xtb, nullptr, hn, alphaB, betaB, nullptr, nullptr,
         DS, H_, CIN, DS, /*tr=*/1, 0, B_ * NC, NC,
         0, (long)DS * CIN, (long)H_ * CIN, 0, (long)NC * sHN, sHN, DS);
    k_ext_head<<<(ROWS + 255) / 256, 256, 0, stream>>>(outExt, hn,
        (const float*)d_in[7], (const float*)d_in[8]);
    k_exist<<<1, 256, 0, stream>>>(exist, outExt);

    // cls GEMM1 + GEMM2 (logits straight into d_out, transposed [h][o])
    gemm(stream, w1B, xtb, nullptr, hn, alphaB + NC * DS, betaB + NC * DS, nullptr, nullptr,
         DS, H_, CIN, DS, 1, 0, B_ * NC, NC,
         0, (long)DS * CIN, (long)H_ * CIN, 0, (long)NC * sHN, sHN, DS);
    gemm(stream, w2c, hn, outCls, nullptr, nullptr, (const float*)d_in[16], nullptr, nullptr,
         W_, H_, DS, W_, 1, 0, B_ * NC, NC,
         0, (long)W_ * DS, (long)NC * sHN, sHN, (long)NC * H_ * W_, (long)H_ * W_, W_);
    k_cls_softmax<<<(ROWS * 32 + 255) / 256, 256, 0, stream>>>(outCls, corr, W_);

    // ============== windows + transformer ==============
    k_build_pad<<<(B_ * DF * H_ * WP + 255) / 256, 256, 0, stream>>>(pad, x);
    k_gather_win<<<(B_ * NC * ITC + 255) / 256, 256, 0, stream>>>(winb, pad, corr);

    gemm(stream, winb, tokwb, tok, nullptr, nullptr, nullptr, tok_b, nullptr,
         B_ * NC, DT, ITC, DT, 0, 0, 1, 1, 0, 0, 0, 0, 0, 0, 0);
    k_add_emb<<<(B_ * NC * DT + 255) / 256, 256, 0, stream>>>(tok, emb);
    k_layernorm<<<B_ * NC, 256, 0, stream>>>(ybf, tok, ln1_g, ln1_b, DT);

    gemm(stream, ybf, qkvwb, qkvb, nullptr, nullptr, nullptr, nullptr, nullptr,
         B_ * NC, 3 * INNER, DT, 3 * INNER, 0, 0, 1, 1, 0, 0, 0, 0, 0, 0, 0);
    k_attn<<<(B_ * HEADS * NC + 255) / 256, 256, 0, stream>>>(attnb, qkvb, exist);

    gemm(stream, attnb, outwb, tok2, nullptr, nullptr, nullptr, out_b, tok,
         B_ * NC, DT, INNER, DT, 0, 0, 1, 1, 0, 0, 0, 0, 0, 0, 0);
    k_layernorm<<<B_ * NC, 256, 0, stream>>>(ybf, tok2, ln2_g, ln2_b, DT);

    gemm(stream, ybf, ffw1b, nullptr, g1b, nullptr, nullptr, ff_b1, nullptr,
         B_ * NC, MLP, DT, MLP, 0, /*gelu=*/1, 1, 1, 0, 0, 0, 0, 0, 0, 0);
    gemm(stream, g1b, ffw2b, tokf, nullptr, nullptr, nullptr, ff_b2, tok2,
         B_ * NC, DT, MLP, DT, 0, 0, 1, 1, 0, 0, 0, 0, 0, 0, 0);
    k_layernorm<<<B_ * NC, 256, 0, stream>>>(ybf, tokf, lnf_g, lnf_b, DT);

    gemm(stream, ybf, decwb, decb, nullptr, nullptr, nullptr, dec_b, nullptr,
         B_ * NC, ITC, DT, ITC, 0, 0, 1, 1, 0, 0, 0, 0, 0, 0, 0);
    k_scatter<<<(B_ * DF * H_ + 255) / 256, 256, 0, stream>>>(pad, decb, corr, exist);

    // ============== Round 2 heads (x2 = pad interior) ==============
    cvt(w1A, (const float*)d_in[17], (size_t)NC * DS * CIN);  // ext2_w1
    cvt(w1B, (const float*)d_in[25], (size_t)NC * DS * CIN);  // cls2_w1
    cvt(w2c, (const float*)d_in[31], (size_t)NC * W_ * DS);   // cls2_w2
    k_build_xt2<<<(B_ * H_ * CIN + 255) / 256, 256, 0, stream>>>(xtb, pad);

    gemm(stream, w1A, xtb, nullptr, hn, alphaB + 2 * NC * DS, betaB + 2 * NC * DS, nullptr, nullptr,
         DS, H_, CIN, DS, 1, 0, B_ * NC, NC,
         0, (long)DS * CIN, (long)H_ * CIN, 0, (long)NC * sHN, sHN, DS);
    k_ext_head<<<(ROWS + 255) / 256, 256, 0, stream>>>(outExt2, hn,
        (const float*)d_in[23], (const float*)d_in[24]);

    gemm(stream, w1B, xtb, nullptr, hn, alphaB + 3 * NC * DS, betaB + 3 * NC * DS, nullptr, nullptr,
         DS, H_, CIN, DS, 1, 0, B_ * NC, NC,
         0, (long)DS * CIN, (long)H_ * CIN, 0, (long)NC * sHN, sHN, DS);
    gemm(stream, w2c, hn, outCls2, nullptr, nullptr, (const float*)d_in[32], nullptr, nullptr,
         W_, H_, DS, W_, 1, 0, B_ * NC, NC,
         0, (long)W_ * DS, (long)NC * sHN, sHN, (long)NC * H_ * W_, (long)H_ * W_, W_);
    k_cls_softmax<<<(ROWS * 32 + 255) / 256, 256, 0, stream>>>(outCls2, corr, W_);
}
